// Memory_57629871178311
// MI455X (gfx1250) — matrix operand order repaired and checked by reading, not compiled, not run on hardware
//
#include <hip/hip_runtime.h>

// ---------------------------------------------------------------------------
// Shapes from the reference
// ---------------------------------------------------------------------------
#define BSZ   1024
#define NMEM  100000
#define KTOP  128
#define CDIM  256
#define ADIM  253
#define DM    256
#define KEY   32
#define QKVD  320
#define NSEG  4
#define SEGLN 25000          // NMEM / NSEG exactly
#define CAP   512            // per-row candidate buffer in top-k kernel

typedef __attribute__((ext_vector_type(16))) __bf16 v16bf;
typedef __attribute__((ext_vector_type(8)))  float  v8f;
typedef __attribute__((ext_vector_type(2)))  float  v2f;

union F8    { v8f  v; float  f[8];  };
union BF16x16 { v16bf v; uint4 u[2]; __bf16 e[16]; };

// ---------------------------------------------------------------------------
// WMMA helpers.
//
// bf16 16x16x32:  A row-major [m][k] (contiguous k), B transposed row-major
// [n][k] (contiguous k).  Fragment layouts per CDNA5 ISA:
//   A lane: m=lane&15, e<8 -> k=kk+(lane>>4)*8+e ; e>=8 -> k=kk+16+(lane>>4)*8+(e-8)
//   B lane: n=lane&15, same k mapping as A
// Both become two 16B loads per fragment.
// ---------------------------------------------------------------------------
__device__ __forceinline__
v8f wmma_bf16_acc(const __bf16* Arow, int lda, const __bf16* Bt, int ldb,
                  int K, int lane, v8f acc)
{
    const int ml = lane & 15, hi = lane >> 4;
    const __bf16* ap = Arow + ml * lda;
    const __bf16* bp = Bt   + ml * ldb;
    for (int kk = 0; kk < K; kk += 32) {
        BF16x16 a, b;
        a.u[0] = *(const uint4*)(ap + kk + hi * 8);
        a.u[1] = *(const uint4*)(ap + kk + 16 + hi * 8);
        b.u[0] = *(const uint4*)(bp + kk + hi * 8);
        b.u[1] = *(const uint4*)(bp + kk + 16 + hi * 8);
        acc = __builtin_amdgcn_wmma_f32_16x16x32_bf16(false, a.v, false, b.v,
                                                      (short)0, acc, false, false);
    }
    return acc;
}

// f32 16x16x4: A row-major [m][k], B transposed row-major [n][k].
//   A lane: m=lane&15, {k0+(lane>>4)*2, +1} ; B lane: n=lane&15, same k pair.
__device__ __forceinline__
v8f wmma_f32_acc(const float* Arow, int lda, const float* Bt, int ldb,
                 int K, int lane, v8f acc)
{
    const int ml = lane & 15, hi = lane >> 4;
    const float* ap = Arow + ml * lda;
    const float* bp = Bt   + ml * ldb;
    for (int k0 = 0; k0 < K; k0 += 4) {
        v2f a = *(const v2f*)(ap + k0 + hi * 2);
        v2f b = *(const v2f*)(bp + k0 + hi * 2);
        acc = __builtin_amdgcn_wmma_f32_16x16x4_f32(false, a, false, b,
                                                    (short)0, acc, false, false);
    }
    return acc;
}

// ---------------------------------------------------------------------------
// Wave-level selection: move the `keep` largest of vb[0..cnt) (value desc,
// index asc tie-break, matching jax.lax.top_k) into vb[0..keep).
// One wave32 executes this; LDS ops from one wave are in-order.
// ---------------------------------------------------------------------------
__device__ void wave_select(volatile float* vb, volatile int* ib,
                            int cnt, int keep, int lane)
{
    for (int j = 0; j < keep; ++j) {
        float bv = -3.4e38f; int bi = 0x7fffffff; int bp = -1;
        for (int t = j + lane; t < cnt; t += 32) {
            float v = vb[t]; int id = ib[t];
            if (v > bv || (v == bv && id < bi)) { bv = v; bi = id; bp = t; }
        }
        for (int off = 16; off; off >>= 1) {
            float ov = __shfl_xor(bv, off, 32);
            int   oi = __shfl_xor(bi, off, 32);
            int   op = __shfl_xor(bp, off, 32);
            if (ov > bv || (ov == bv && oi < bi)) { bv = ov; bi = oi; bp = op; }
        }
        if (lane == 0 && bp >= 0 && bp != j) {
            float tv = vb[j]; int ti = ib[j];
            vb[j] = bv; ib[j] = bi; vb[bp] = tv; ib[bp] = ti;
        }
        __builtin_amdgcn_wave_barrier();
        asm volatile("s_wait_dscnt 0" ::: "memory");
    }
}

// ---------------------------------------------------------------------------
// Kernel 1: cn = c / ||c||   (one block per row)
// ---------------------------------------------------------------------------
__global__ void k_norm_c(const float* __restrict__ c, float* __restrict__ cn)
{
    __shared__ float red[256];
    int b = blockIdx.x, t = threadIdx.x;
    float x = c[b * CDIM + t];
    red[t] = x * x; __syncthreads();
    for (int s = 128; s > 0; s >>= 1) { if (t < s) red[t] += red[t + s]; __syncthreads(); }
    cn[b * CDIM + t] = x * rsqrtf(red[0] + 1e-12f);
}

// Kernel 2: minv[n] = 1/||mem_c[n]||
__global__ void k_minv(const float* __restrict__ mc, float* __restrict__ minv)
{
    __shared__ float red[256];
    int b = blockIdx.x, t = threadIdx.x;
    float x = mc[(size_t)b * CDIM + t];
    red[t] = x * x; __syncthreads();
    for (int s = 128; s > 0; s >>= 1) { if (t < s) red[t] += red[t + s]; __syncthreads(); }
    if (t == 0) minv[b] = rsqrtf(red[0] + 1e-12f);
}

// Kernel 3: transpose GEMM weights to bf16 [n][k] so WMMA B fragments are
// contiguous 16B loads.
__global__ void k_prep_w(const float* __restrict__ qkv_w, const float* __restrict__ w1,
                         const float* __restrict__ w2,
                         __bf16* __restrict__ qkvT, __bf16* __restrict__ w1T,
                         __bf16* __restrict__ w2T)
{
    int i = blockIdx.x * 256 + threadIdx.x;
    if (i < QKVD * DM) {
        int n = i / DM, k = i - n * DM;
        qkvT[n * DM + k] = (__bf16)qkv_w[k * QKVD + n];
    } else if (i < QKVD * DM + DM * DM) {
        int j = i - QKVD * DM; int n = j / DM, k = j - n * DM;
        w1T[n * DM + k] = (__bf16)w1[k * DM + n];
    } else if (i < QKVD * DM + 2 * DM * DM) {
        int j = i - QKVD * DM - DM * DM; int n = j / DM, k = j - n * DM;
        w2T[n * DM + k] = (__bf16)w2[k * DM + n];
    }
}

// ---------------------------------------------------------------------------
// Kernel 4: fused cosine-similarity GEMM (f32 WMMA) + streaming top-128.
// grid = (B/16, NSEG); block = 512 (16 waves).  Wave w computes a 16x16
// similarity tile per chunk; candidates above the row threshold are appended
// to a per-row LDS buffer and periodically compacted to top-128 by one wave.
// ---------------------------------------------------------------------------
__global__ void __launch_bounds__(512)
k_sim_topk(const float* __restrict__ cn, const float* __restrict__ mem_c,
           const float* __restrict__ minv,
           float* __restrict__ stage_v, int* __restrict__ stage_i)
{
    extern __shared__ unsigned char smem[];
    float* cnT  = (float*)smem;                        // 16*256 f32
    float* vbuf = (float*)(smem + 16384);              // 16*CAP f32
    int*   ibuf = (int*)  (smem + 16384 + 32768);      // 16*CAP i32
    int*   cnt  = (int*)  (smem + 16384 + 65536);      // 16
    float* thr  = (float*)(smem + 16384 + 65536 + 64); // 16

    const int b0 = blockIdx.x * 16;
    const int seg = blockIdx.y;
    const int segBase = seg * SEGLN, segEnd = segBase + SEGLN;
    const int TID = threadIdx.x, w = TID >> 5, lane = TID & 31;
    const int ml = lane & 15, hi = lane >> 4;

    for (int i = TID; i < 16 * CDIM; i += 512) cnT[i] = cn[b0 * CDIM + i];
    if (TID < 16) { cnt[TID] = 0; thr[TID] = -3.4e38f; }
    __syncthreads();

    const int chunks = (SEGLN + 255) / 256;
    for (int ch = 0; ch < chunks; ++ch) {
        int n = segBase + ch * 256 + w * 16 + ml;
        bool valid = (n < segEnd);
        const float* brow = mem_c + (size_t)(valid ? n : segBase) * CDIM;
        const float* arow = cnT + ml * CDIM;
        v8f acc = {};
        for (int k0 = 0; k0 < CDIM; k0 += 4) {
            v2f a = *(const v2f*)(arow + k0 + hi * 2);
            v2f b = *(const v2f*)(brow + k0 + hi * 2);
            acc = __builtin_amdgcn_wmma_f32_16x16x4_f32(false, a, false, b,
                                                        (short)0, acc, false, false);
        }
        float sc = valid ? minv[n] : 0.f;
        F8 a8; a8.v = acc;
        for (int v = 0; v < 8; ++v) {
            int r = v + 8 * hi;
            float val = a8.f[v] * sc;
            if (valid && val > thr[r]) {
                int pos = atomicAdd(&cnt[r], 1);
                if (pos < CAP) { vbuf[r * CAP + pos] = val; ibuf[r * CAP + pos] = n; }
            }
        }
        __syncthreads();
        if (w < 16 && cnt[w] > CAP - 256) {     // compact before overflow possible
            int c2 = cnt[w] > CAP ? CAP : cnt[w];
            wave_select(vbuf + w * CAP, ibuf + w * CAP, c2, KTOP, lane);
            if (lane == 0) { cnt[w] = KTOP; thr[w] = vbuf[w * CAP + KTOP - 1]; }
        }
        __syncthreads();
    }

    if (w < 16) {
        int c2 = cnt[w] > CAP ? CAP : cnt[w];
        int keep = c2 < KTOP ? c2 : KTOP;
        wave_select(vbuf + w * CAP, ibuf + w * CAP, c2, keep, lane);
        for (int j = lane; j < KTOP; j += 32) {
            size_t o = ((size_t)seg * BSZ + (b0 + w)) * KTOP + j;
            stage_v[o] = (j < keep) ? vbuf[w * CAP + j] : -3.4e38f;
            stage_i[o] = (j < keep) ? ibuf[w * CAP + j] : 0;
        }
    }
}

// Kernel 5: merge NSEG partial top-128 lists (disjoint index ranges) -> final.
__global__ void k_merge(const float* __restrict__ stage_v, const int* __restrict__ stage_i,
                        float* __restrict__ topv, int* __restrict__ topi)
{
    __shared__ float vb[NSEG * KTOP];
    __shared__ int   ib[NSEG * KTOP];
    int b = blockIdx.x, lane = threadIdx.x;
    for (int i = lane; i < NSEG * KTOP; i += 32) {
        int seg = i >> 7, j = i & (KTOP - 1);
        size_t o = ((size_t)seg * BSZ + b) * KTOP + j;
        vb[i] = stage_v[o]; ib[i] = stage_i[o];
    }
    __syncthreads();
    wave_select(vb, ib, NSEG * KTOP, KTOP, lane);
    for (int j = lane; j < KTOP; j += 32) {
        topv[b * KTOP + j] = vb[j];
        topi[b * KTOP + j] = ib[j];
    }
}

// Kernel 6: metadata = [top_d, mem_q[idx], mem_t[idx], action]  (B,128,256)
__global__ void k_gather(const float* __restrict__ topv, const int* __restrict__ topi,
                         const float* __restrict__ mem_q, const float* __restrict__ mem_t,
                         const float* __restrict__ action, float* __restrict__ meta)
{
    int bid = blockIdx.x;              // b*128 + j
    int b = bid >> 7, j = bid & (KTOP - 1);
    int d = threadIdx.x;
    float val;
    if      (d == 0) val = topv[b * KTOP + j];
    else if (d == 1) val = mem_q[topi[b * KTOP + j]];
    else if (d == 2) val = mem_t[topi[b * KTOP + j]];
    else             val = action[b * ADIM + (d - 3)];
    meta[((size_t)b * KTOP + j) * DM + d] = val;
}

// ---------------------------------------------------------------------------
// Kernel 7: per-batch token block (one workgroup per batch element, 16 waves)
//   qkv = LN(meta@Wqkv + b);  q,k,v split;  scores f32-WMMA + softmax;
//   att bf16-WMMA;  mem = LN(meta+att);  MLP (bf16-WMMA) with residual LN;
//   mean over tokens; 2-layer projection.
// LN stats via deterministic per-wave shuffle reductions into s_stats.
// ---------------------------------------------------------------------------
__device__ __forceinline__
void rowstats(float* s_stats, float* s_row, int TID, float invD)
{
    if (TID < 128) {
        int mtt = TID >> 4, lr = TID & 15;
        int i0 = ((mtt * 2) * 16 + lr) * 2, i1 = ((mtt * 2 + 1) * 16 + lr) * 2;
        float s1 = s_stats[i0] + s_stats[i1];
        float s2 = s_stats[i0 + 1] + s_stats[i1 + 1];
        float mean = s1 * invD;
        float var  = s2 * invD - mean * mean;
        s_row[TID * 2]     = mean;
        s_row[TID * 2 + 1] = rsqrtf(var + 1e-5f);
    }
}

__global__ void __launch_bounds__(512)
k_token(const float* __restrict__ meta_all,
        const __bf16* __restrict__ qkvT, const __bf16* __restrict__ w1T,
        const __bf16* __restrict__ w2T,
        const float* __restrict__ qkv_b, const float* __restrict__ ln_g,
        const float* __restrict__ ln_b,  const float* __restrict__ lnm_g,
        const float* __restrict__ lnm_b, const float* __restrict__ b1,
        const float* __restrict__ b2,    const float* __restrict__ p1w,
        const float* __restrict__ p1b,   const float* __restrict__ p2w,
        const float* __restrict__ p2b,   float* __restrict__ out)
{
    extern __shared__ unsigned char smem[];
    __bf16* s_abf = (__bf16*)smem;                       // 128*256 bf16: meta -> v_t -> h1
    __bf16* s_obf = (__bf16*)(smem + 65536);             // 128*320 bf16: qkv -> att -> mem
    unsigned char* p2m = smem + 65536 + 81920;
    float*  s_sc  = (float*)p2m;                         // 128*128 f32 scores
    __bf16* s_y   = (__bf16*)p2m;                        //   (aliased) 128*256 bf16 y
    float*  s_qk  = (float*)(p2m + 65536);               // 128*64 f32  q||k
    float*  s_stats = (float*)(p2m + 65536 + 32768);     // 16 waves * 16 rows * 2
    float*  s_row   = s_stats + 512;                     // 128 * {mean, invstd}
    float*  s_pool  = s_row + 256;                       // 256 pooled + 256 hidden

    const int b = blockIdx.x;
    const int TID = threadIdx.x, w = TID >> 5, lane = TID & 31;
    const int ml = lane & 15, hi = lane >> 4;
    const int mt = w >> 1, half = w & 1, m0 = mt * 16;
    const float* meta_g = meta_all + (size_t)b * (KTOP * DM);

    // P0: meta -> bf16 A operand
    for (int i = TID; i < KTOP * DM; i += 512) s_abf[i] = (__bf16)meta_g[i];
    s_stats[TID] = 0.f;
    __syncthreads();

    // P1: qkv = meta @ Wqkv + b  (raw to s_obf bf16, stats in f32)
    for (int t = 0; t < 10; ++t) {
        int n0 = (half * 10 + t) * 16;
        v8f acc = {};
        acc = wmma_bf16_acc(s_abf + m0 * DM, DM, qkvT + n0 * DM, DM, DM, lane, acc);
        float bias = qkv_b[n0 + ml];
        F8 a; a.v = acc;
        for (int v = 0; v < 8; ++v) {
            int lr = v + 8 * hi;
            float val = a.f[v] + bias;
            s_obf[(m0 + lr) * QKVD + n0 + ml] = (__bf16)val;
            float s1 = val, s2 = val * val;
            for (int off = 1; off < 16; off <<= 1) {
                s1 += __shfl_xor(s1, off, 32); s2 += __shfl_xor(s2, off, 32);
            }
            if (ml == 0) {
                s_stats[(w * 16 + lr) * 2]     += s1;
                s_stats[(w * 16 + lr) * 2 + 1] += s2;
            }
        }
    }
    __syncthreads();
    rowstats(s_stats, s_row, TID, 1.f / (float)QKVD);
    __syncthreads();

    // P2: LN + split q (scaled), k, v (transposed [d][m] bf16 into s_abf)
    for (int i = TID; i < KTOP * QKVD; i += 512) {
        int m = i / QKVD, j = i - m * QKVD;
        float x  = (float)s_obf[i];
        float xn = (x - s_row[m * 2]) * s_row[m * 2 + 1] * ln_g[j] + ln_b[j];
        if      (j < KEY)     s_qk[m * 64 + j] = xn * 0.17677669529663687f; // KEY^-0.5
        else if (j < 2 * KEY) s_qk[m * 64 + j] = xn;                        // slots 32..63
        else                  s_abf[(j - 2 * KEY) * KTOP + m] = (__bf16)xn; // v_t
    }
    __syncthreads();

    // P3: scores = q @ k^T  (f32 WMMA, K=32) -> softmax -> att bf16 in s_obf
    for (int t = 0; t < 4; ++t) {
        int n0 = (half * 4 + t) * 16;
        v8f acc = {};
        acc = wmma_f32_acc(s_qk + m0 * 64, 64, s_qk + KEY + n0 * 64, 64, KEY, lane, acc);
        F8 a; a.v = acc;
        for (int v = 0; v < 8; ++v) s_sc[(m0 + v + 8 * hi) * KTOP + n0 + ml] = a.f[v];
    }
    __syncthreads();
    for (int q8 = 0; q8 < 8; ++q8) {
        int r = w * 8 + q8;
        float vr[4], mx = -3.4e38f;
        for (int i = 0; i < 4; ++i) { vr[i] = s_sc[r * KTOP + lane + i * 32]; mx = fmaxf(mx, vr[i]); }
        for (int off = 16; off; off >>= 1) mx = fmaxf(mx, __shfl_xor(mx, off, 32));
        float sum = 0.f;
        for (int i = 0; i < 4; ++i) { vr[i] = __expf(vr[i] - mx); sum += vr[i]; }
        for (int off = 16; off; off >>= 1) sum += __shfl_xor(sum, off, 32);
        float inv = 1.f / sum;
        for (int i = 0; i < 4; ++i)
            s_obf[r * KTOP + lane + i * 32] = (__bf16)(vr[i] * inv);
    }
    __syncthreads();
    s_stats[TID] = 0.f;
    __syncthreads();

    // P4: y = att@v + meta  (bf16 WMMA, K=128; B = v_t contiguous)
    for (int t = 0; t < 8; ++t) {
        int n0 = (half * 8 + t) * 16;
        v8f acc = {};
        acc = wmma_bf16_acc(s_obf + m0 * KTOP, KTOP, s_abf + n0 * KTOP, KTOP, KTOP, lane, acc);
        F8 a; a.v = acc;
        for (int v = 0; v < 8; ++v) {
            int lr = v + 8 * hi, m = m0 + lr, n = n0 + ml;
            float y = a.f[v] + meta_g[m * DM + n];
            s_y[m * DM + n] = (__bf16)y;
            float s1 = y, s2 = y * y;
            for (int off = 1; off < 16; off <<= 1) {
                s1 += __shfl_xor(s1, off, 32); s2 += __shfl_xor(s2, off, 32);
            }
            if (ml == 0) {
                s_stats[(w * 16 + lr) * 2]     += s1;
                s_stats[(w * 16 + lr) * 2 + 1] += s2;
            }
        }
    }
    __syncthreads();
    rowstats(s_stats, s_row, TID, 1.f / (float)DM);
    __syncthreads();

    // P5: mem = LN(y) -> bf16 in s_obf (ld 256)
    for (int i = TID; i < KTOP * DM; i += 512) {
        int m = i >> 8, j = i & 255;
        float x = (float)s_y[i];
        s_obf[i] = (__bf16)((x - s_row[m * 2]) * s_row[m * 2 + 1] * lnm_g[j] + lnm_b[j]);
    }
    __syncthreads();

    // P6: h1 = relu(mem@W1 + b1) -> s_abf bf16
    for (int t = 0; t < 8; ++t) {
        int n0 = (half * 8 + t) * 16;
        v8f acc = {};
        acc = wmma_bf16_acc(s_obf + m0 * DM, DM, w1T + n0 * DM, DM, DM, lane, acc);
        float bias = b1[n0 + ml];
        F8 a; a.v = acc;
        for (int v = 0; v < 8; ++v) {
            int m = m0 + v + 8 * hi;
            s_abf[m * DM + n0 + ml] = (__bf16)fmaxf(a.f[v] + bias, 0.f);
        }
    }
    __syncthreads();
    s_stats[TID] = 0.f;
    __syncthreads();

    // P7: y2 = h1@W2 + b2 + mem  (stats for final LN)
    for (int t = 0; t < 8; ++t) {
        int n0 = (half * 8 + t) * 16;
        v8f acc = {};
        acc = wmma_bf16_acc(s_abf + m0 * DM, DM, w2T + n0 * DM, DM, DM, lane, acc);
        float bias = b2[n0 + ml];
        F8 a; a.v = acc;
        for (int v = 0; v < 8; ++v) {
            int lr = v + 8 * hi, m = m0 + lr, n = n0 + ml;
            float y2 = a.f[v] + bias + (float)s_obf[m * DM + n];
            s_y[m * DM + n] = (__bf16)y2;
            float s1 = y2, s2 = y2 * y2;
            for (int off = 1; off < 16; off <<= 1) {
                s1 += __shfl_xor(s1, off, 32); s2 += __shfl_xor(s2, off, 32);
            }
            if (ml == 0) {
                s_stats[(w * 16 + lr) * 2]     += s1;
                s_stats[(w * 16 + lr) * 2 + 1] += s2;
            }
        }
    }
    __syncthreads();
    rowstats(s_stats, s_row, TID, 1.f / (float)DM);
    __syncthreads();

    // P8: pooled = mean_m LN(y2);  two small GEMVs for the projection head
    if (TID < DM) {
        float acc = 0.f;
        for (int m = 0; m < KTOP; ++m) {
            float x = (float)s_y[m * DM + TID];
            acc += (x - s_row[m * 2]) * s_row[m * 2 + 1];
        }
        s_pool[TID] = (acc * (1.f / (float)KTOP)) * lnm_g[TID] + lnm_b[TID];
    }
    __syncthreads();
    if (TID < DM) {
        float h = p1b[TID];
        for (int e = 0; e < DM; ++e) h += s_pool[e] * p1w[e * DM + TID];
        s_pool[DM + TID] = fmaxf(h, 0.f);
    }
    __syncthreads();
    if (TID < CDIM) {
        float o = p2b[TID];
        for (int d = 0; d < DM; ++d) o += s_pool[DM + d] * p2w[d * CDIM + TID];
        out[b * CDIM + TID] = o;
    }
}

// ---------------------------------------------------------------------------
// Host launcher
// ---------------------------------------------------------------------------
extern "C" void kernel_launch(void* const* d_in, const int* in_sizes, int n_in,
                              void* d_out, int out_size, void* d_ws, size_t ws_size,
                              hipStream_t stream)
{
    (void)in_sizes; (void)n_in; (void)out_size; (void)ws_size;
    const float* c      = (const float*)d_in[0];
    const float* action = (const float*)d_in[1];
    const float* mem_c  = (const float*)d_in[2];
    const float* mem_q  = (const float*)d_in[3];
    const float* mem_t  = (const float*)d_in[4];
    const float* qkv_w  = (const float*)d_in[5];
    const float* qkv_b  = (const float*)d_in[6];
    const float* ln_g   = (const float*)d_in[7];
    const float* ln_b   = (const float*)d_in[8];
    const float* lnm_g  = (const float*)d_in[9];
    const float* lnm_b  = (const float*)d_in[10];
    const float* w1     = (const float*)d_in[11];
    const float* b1     = (const float*)d_in[12];
    const float* w2     = (const float*)d_in[13];
    const float* b2     = (const float*)d_in[14];
    const float* p1w    = (const float*)d_in[15];
    const float* p1b    = (const float*)d_in[16];
    const float* p2w    = (const float*)d_in[17];
    const float* p2b    = (const float*)d_in[18];
    float* out = (float*)d_out;

    size_t off = 0;
    auto carve = [&](size_t bytes) -> void* {
        void* p = (void*)((char*)d_ws + off);
        off = (off + bytes + 255) & ~(size_t)255;
        return p;
    };
    float*  cn      = (float*)carve((size_t)BSZ * CDIM * 4);
    float*  minv    = (float*)carve((size_t)NMEM * 4);
    float*  stage_v = (float*)carve((size_t)NSEG * BSZ * KTOP * 4);
    int*    stage_i = (int*)  carve((size_t)NSEG * BSZ * KTOP * 4);
    float*  topv    = (float*)carve((size_t)BSZ * KTOP * 4);
    int*    topi    = (int*)  carve((size_t)BSZ * KTOP * 4);
    __bf16* qkvT    = (__bf16*)carve((size_t)QKVD * DM * 2);
    __bf16* w1T     = (__bf16*)carve((size_t)DM * DM * 2);
    __bf16* w2T     = (__bf16*)carve((size_t)DM * DM * 2);
    float*  meta    = (float*)carve((size_t)BSZ * KTOP * DM * 4);

    k_norm_c<<<BSZ, 256, 0, stream>>>(c, cn);
    k_minv  <<<NMEM, 256, 0, stream>>>(mem_c, minv);
    k_prep_w<<<(QKVD * DM + 2 * DM * DM + 255) / 256, 256, 0, stream>>>(
        qkv_w, w1, w2, qkvT, w1T, w2T);

    const size_t smem3 = 16384 + 32768 + 32768 + 64 + 64;     // ~82 KB
    k_sim_topk<<<dim3(BSZ / 16, NSEG), 512, smem3, stream>>>(
        cn, mem_c, minv, stage_v, stage_i);

    k_merge<<<BSZ, 32, 0, stream>>>(stage_v, stage_i, topv, topi);

    k_gather<<<BSZ * KTOP, 256, 0, stream>>>(topv, topi, mem_q, mem_t, action, meta);

    const size_t smem4 = 65536 + 81920 + 65536 + 32768 + 512 * 4 + 256 * 4 + 512 * 4;
    k_token<<<BSZ, 512, smem4, stream>>>(
        meta, qkvT, w1T, w2T, qkv_b, ln_g, ln_b, lnm_g, lnm_b,
        b1, b2, p1w, p1b, p2w, p2b, out);
}
